// SupervisedConLoss_70781061038765
// MI455X (gfx1250) — matrix operand order repaired and checked by reading, not compiled, not run on hardware
//
#include <hip/hip_runtime.h>
#include <hip/hip_bf16.h>

// SupCon loss for MI455X (gfx1250). Heavy part: F (8192x128 fp32) times F^T.
// Primary path: split-precision bf16x3 GEMM on V_WMMA_F32_16X16X32_BF16
// (fp32-class accuracy, 8.5x fewer matrix-pipe cycles than f32 K=4 WMMA,
// co-executes with the VALU softmax). Fallback path: V_WMMA_F32_16X16X4_F32.
// Online softmax: the 8192x8192 logits matrix is never materialized.

#define BS 8192
#define DIM 128
#define NUM_CLASSES 10
#define TEMP 0.1f
#define BASE_TEMPERATURE 0.07f
#define NTILES (BS / 16)   // 512
#define KCHUNKS (DIM / 4)  // 32 (f32 path)
#define KCH32 (DIM / 32)   // 4  (bf16 path)

typedef __attribute__((ext_vector_type(2)))  float  v2f;
typedef __attribute__((ext_vector_type(8)))  float  v8f;
typedef __attribute__((ext_vector_type(8)))  __bf16 v8bf;
typedef __attribute__((ext_vector_type(16))) __bf16 v16bf;

// ---------------- workspace layout (float units) ----------------
#define WS_CLSSUM   0          // 10*128 floats
#define WS_LOSSSUM  1280       // 1 float
#define WS_COUNTS   1282       // 10 ints
#define WS_ROWM     1312       // 2 * BS * nsplit floats (partial m,s)
#define WS_FHI      132384     // BS*DIM bf16  (= 524288 floats) ; 32B aligned
#define WS_FLO      (WS_FHI + (BS * DIM) / 2)

// ---------------- kernel 1: zero init ----------------
__global__ void scl_zero(float* ws) {
    int i = blockIdx.x * 256 + threadIdx.x;
    if (i <= WS_LOSSSUM) ws[i] = 0.0f;
}

// ---------------- kernel 1b: split F*(1/TEMP) into bf16 hi + lo ----------------
__global__ __launch_bounds__(256) void scl_split(const float* __restrict__ F,
                                                 __bf16* __restrict__ Fhi,
                                                 __bf16* __restrict__ Flo) {
    int i = blockIdx.x * 256 + threadIdx.x;
    if (i < BS * DIM) {
        float x = F[i] * (1.0f / TEMP);
        __bf16 h = (__bf16)x;
        float r = x - (float)h;        // exact in f32
        Fhi[i] = h;
        Flo[i] = (__bf16)r;
    }
}

// ---------------- kernel 2: per-class feature sums ----------------
__global__ __launch_bounds__(128) void scl_classsum(const float* __restrict__ F,
                                                    const int* __restrict__ T,
                                                    float* __restrict__ clsSum) {
    __shared__ float ls[NUM_CLASSES * DIM];
    const int col = threadIdx.x;
    #pragma unroll
    for (int k = 0; k < NUM_CLASSES; ++k) ls[k * DIM + col] = 0.0f;
    __syncthreads();
    const int r0 = blockIdx.x * 64;
    for (int r = 0; r < 64; ++r) {
        const int row = r0 + r;
        const int cls = T[row];
        ls[cls * DIM + col] += F[row * DIM + col];
    }
    __syncthreads();
    #pragma unroll
    for (int k = 0; k < NUM_CLASSES; ++k)
        atomicAdd(&clsSum[k * DIM + col], ls[k * DIM + col]);
}

// ---------------- kernel 3: class histogram ----------------
__global__ __launch_bounds__(256) void scl_counts(const int* __restrict__ T,
                                                  int* __restrict__ counts) {
    __shared__ int lc[NUM_CLASSES];
    if (threadIdx.x < NUM_CLASSES) lc[threadIdx.x] = 0;
    __syncthreads();
    for (int i = threadIdx.x; i < BS; i += 256) atomicAdd(&lc[T[i]], 1);
    __syncthreads();
    if (threadIdx.x < NUM_CLASSES) counts[threadIdx.x] = lc[threadIdx.x];
}

// ---------------- shared device helpers ----------------
__device__ __forceinline__ void online_update(float* rm, float* rs, const v8f& acc,
                                              int m0, int n0, int half, int lrow) {
    const int gn = n0 + lrow;
    #pragma unroll
    for (int r = 0; r < 8; ++r) {
        const float x = acc[r];                  // logits[gm][gn] (already /TEMP)
        const int gm = m0 + r + 8 * half;
        const float mn = fmaxf(rm[r], x);
        const float e = (gm == gn) ? 0.0f : __expf(x - mn);  // mask diagonal
        rs[r] = rs[r] * __expf(rm[r] - mn) + e;
        rm[r] = mn;
    }
}

__device__ __forceinline__ void merge_and_store(float* rm, float* rs, int m0, int half,
                                                int lrow, int chunk, int nsplit,
                                                float* rowM, float* rowS) {
    #pragma unroll
    for (int r = 0; r < 8; ++r) {
        float m = rm[r], s = rs[r];
        #pragma unroll
        for (int mask = 1; mask < 16; mask <<= 1) {
            const float mo = __shfl_xor(m, mask, 32);
            const float so = __shfl_xor(s, mask, 32);
            const float mn = fmaxf(m, mo);
            s = s * __expf(m - mn) + so * __expf(mo - mn);
            m = mn;
        }
        if (lrow == 0) {
            const int row = m0 + r + 8 * half;
            rowM[row * nsplit + chunk] = m;
            rowS[row * nsplit + chunk] = s;
        }
    }
}

// ---------------- kernel 4a: bf16x3 WMMA GEMM + online softmax ----------------
// A 16-bit 16x32 fragment layout (matches B): lane half h -> K=[16h,16h+16),
// loaded as two 16B runs [16h,16h+8) and [16h+8,16h+16).
// B 16-bit 32x16 fragment (contiguous k order, same as A): lane half h -> K=[16h,16h+16),
// one contiguous 32B run per K-chunk.
__global__ __launch_bounds__(128) void scl_gemm_softmax_bf16(
        const __bf16* __restrict__ Fhi, const __bf16* __restrict__ Flo,
        float* __restrict__ rowM, float* __restrict__ rowS, int nsplit) {
    const int lane = threadIdx.x & 31;
    const int wave = threadIdx.x >> 5;
    const int half = lane >> 4;
    const int lrow = lane & 15;
    const int gwave = blockIdx.x * 4 + wave;
    const int mtile = gwave / nsplit;
    const int chunk = gwave % nsplit;
    const int tilesPerChunk = NTILES / nsplit;
    const int m0 = mtile * 16;

    // A fragments (hi and lo), held for all 4 K-chunks: 2 x 32 VGPRs
    v16bf ahi[KCH32], alo[KCH32];
    {
        const __bf16* Ah = Fhi + (size_t)(m0 + lrow) * DIM + 16 * half;
        const __bf16* Al = Flo + (size_t)(m0 + lrow) * DIM + 16 * half;
        #pragma unroll
        for (int c = 0; c < KCH32; ++c) {
            union { v16bf v; v8bf h[2]; } u;
            u.h[0] = *(const v8bf*)(Ah + 32 * c);
            u.h[1] = *(const v8bf*)(Ah + 32 * c + 8);
            ahi[c] = u.v;
            u.h[0] = *(const v8bf*)(Al + 32 * c);
            u.h[1] = *(const v8bf*)(Al + 32 * c + 8);
            alo[c] = u.v;
        }
    }

    float rm[8], rs[8];
    #pragma unroll
    for (int r = 0; r < 8; ++r) { rm[r] = -3.0e38f; rs[r] = 0.0f; }

    const int ntBeg = chunk * tilesPerChunk;
    const int ntEnd = ntBeg + tilesPerChunk;

    {   // warm up first tile (each bf16 tile = 16*128*2B = 4KB contiguous)
        const char* p1 = (const char*)(Fhi + (size_t)ntBeg * 16 * DIM);
        const char* p2 = (const char*)(Flo + (size_t)ntBeg * 16 * DIM);
        __builtin_prefetch(p1 + lane * 128, 0, 3);
        __builtin_prefetch(p2 + lane * 128, 0, 3);
    }

    for (int nt = ntBeg; nt < ntEnd; ++nt) {
        const int n0 = nt * 16;
        const __bf16* Bh = Fhi + (size_t)(n0 + lrow) * DIM + 16 * half;
        const __bf16* Bl = Flo + (size_t)(n0 + lrow) * DIM + 16 * half;

        if (nt + 1 < ntEnd) {
            const char* p1 = (const char*)(Fhi + (size_t)(nt + 1) * 16 * DIM);
            const char* p2 = (const char*)(Flo + (size_t)(nt + 1) * 16 * DIM);
            __builtin_prefetch(p1 + lane * 128, 0, 3);
            __builtin_prefetch(p2 + lane * 128, 0, 3);
        }

        // batch-load all B fragments for this tile (keeps loads clustered)
        v16bf bh[KCH32], bl[KCH32];
        #pragma unroll
        for (int c = 0; c < KCH32; ++c) {
            bh[c] = *(const v16bf*)(Bh + 32 * c);
            bl[c] = *(const v16bf*)(Bl + 32 * c);
        }

        v8f acc = {};
        #pragma unroll
        for (int c = 0; c < KCH32; ++c) {
            acc = __builtin_amdgcn_wmma_f32_16x16x32_bf16(
                false, ahi[c], false, bh[c], (short)0, acc, false, false);
            acc = __builtin_amdgcn_wmma_f32_16x16x32_bf16(
                false, ahi[c], false, bl[c], (short)0, acc, false, false);
            acc = __builtin_amdgcn_wmma_f32_16x16x32_bf16(
                false, alo[c], false, bh[c], (short)0, acc, false, false);
        }

        online_update(rm, rs, acc, m0, n0, half, lrow);
    }

    merge_and_store(rm, rs, m0, half, lrow, chunk, nsplit, rowM, rowS);
}

// ---------------- kernel 4b: f32 WMMA fallback (small ws_size) ----------------
__global__ __launch_bounds__(128) void scl_gemm_softmax_f32(
        const float* __restrict__ F,
        float* __restrict__ rowM, float* __restrict__ rowS, int nsplit) {
    const int lane = threadIdx.x & 31;
    const int wave = threadIdx.x >> 5;
    const int half = lane >> 4;
    const int lrow = lane & 15;
    const int gwave = blockIdx.x * 4 + wave;
    const int mtile = gwave / nsplit;
    const int chunk = gwave % nsplit;
    const int tilesPerChunk = NTILES / nsplit;
    const int m0 = mtile * 16;
    const float invT = 1.0f / TEMP;

    v2f a[KCHUNKS];
    {
        const float* Ap = F + (m0 + lrow) * DIM + 2 * half;
        #pragma unroll
        for (int c = 0; c < KCHUNKS; ++c) {
            v2f t = *(const v2f*)(Ap + 4 * c);
            a[c] = t * invT;
        }
    }

    float rm[8], rs[8];
    #pragma unroll
    for (int r = 0; r < 8; ++r) { rm[r] = -3.0e38f; rs[r] = 0.0f; }

    const int ntBeg = chunk * tilesPerChunk;
    const int ntEnd = ntBeg + tilesPerChunk;

    for (int nt = ntBeg; nt < ntEnd; ++nt) {
        const int n0 = nt * 16;
        const float* Bp = F + (n0 + lrow) * DIM + 2 * half;
        if (nt + 1 < ntEnd) {
            const char* nb = (const char*)(F + (size_t)(nt + 1) * 16 * DIM);
            __builtin_prefetch(nb + lane * 256, 0, 3);
        }
        v2f b[KCHUNKS];
        #pragma unroll
        for (int c = 0; c < KCHUNKS; ++c)
            b[c] = *(const v2f*)(Bp + 4 * c);
        v8f acc = {};
        #pragma unroll
        for (int c = 0; c < KCHUNKS; ++c)
            acc = __builtin_amdgcn_wmma_f32_16x16x4_f32(
                false, a[c], false, b[c], (short)0, acc, false, false);
        online_update(rm, rs, acc, m0, n0, half, lrow);
    }

    merge_and_store(rm, rs, m0, half, lrow, chunk, nsplit, rowM, rowS);
}

// ---------------- kernel 5: per-row finalize ----------------
__global__ __launch_bounds__(256) void scl_rowloss(const float* __restrict__ F,
                                                   const int* __restrict__ T,
                                                   const float* __restrict__ clsSum,
                                                   const int* __restrict__ counts,
                                                   const float* __restrict__ rowM,
                                                   const float* __restrict__ rowS,
                                                   float* __restrict__ lossSum,
                                                   int nsplit) {
    const int lane = threadIdx.x & 31;
    const int wave = threadIdx.x >> 5;
    const int row = blockIdx.x * 8 + wave;
    const int cls = T[row];
    const float* f  = F + row * DIM;
    const float* cs = clsSum + cls * DIM;

    float dotc = 0.0f, dots = 0.0f;
    #pragma unroll
    for (int k = lane; k < DIM; k += 32) {
        const float v = f[k];
        dotc += v * cs[k];
        dots += v * v;
    }
    #pragma unroll
    for (int mask = 16; mask >= 1; mask >>= 1) {
        dotc += __shfl_xor(dotc, mask, 32);
        dots += __shfl_xor(dots, mask, 32);
    }

    if (lane == 0) {
        float m = -3.0e38f, s = 0.0f;
        for (int c = 0; c < nsplit; ++c) {
            const float mo = rowM[row * nsplit + c];
            const float so = rowS[row * nsplit + c];
            const float mn = fmaxf(m, mo);
            s = s * __expf(m - mn) + so * __expf(mo - mn);
            m = mn;
        }
        const float npos   = (float)(counts[cls] - 1);
        const float possum = (dotc - dots) * (1.0f / TEMP);
        const float lse    = m + __logf(s + 1e-20f);
        const float mlp    = (possum - npos * lse) / (npos + 1e-20f);
        atomicAdd(lossSum, mlp);
    }
}

// ---------------- kernel 6: outputs ----------------
__global__ void scl_final(const int* __restrict__ counts,
                          const float* __restrict__ lossSum,
                          float* __restrict__ out) {
    if (threadIdx.x == 0) {
        float pos = 0.0f, neg = 0.0f;
        #pragma unroll
        for (int k = 0; k < NUM_CLASSES; ++k) {
            const float c = (float)counts[k];
            pos += c * (c - 1.0f);
            neg += c * ((float)BS - c);
        }
        out[0] = -(TEMP / BASE_TEMPERATURE) * (*lossSum) / (float)BS;
        out[1] = pos / (float)BS;
        out[2] = neg / (float)BS;
    }
}

extern "C" void kernel_launch(void* const* d_in, const int* in_sizes, int n_in,
                              void* d_out, int out_size, void* d_ws, size_t ws_size,
                              hipStream_t stream) {
    (void)in_sizes; (void)n_in; (void)out_size;
    const float* F = (const float*)d_in[0];
    const int*   T = (const int*)d_in[1];
    float* ws      = (float*)d_ws;
    float* out     = (float*)d_out;

    float* clsSum   = ws + WS_CLSSUM;
    float* lossSum  = ws + WS_LOSSSUM;
    int*   counts   = (int*)(ws + WS_COUNTS);

    // deterministic path choice from fixed ws_size
    const size_t needBf16 = (size_t)WS_FHI * 4 + 2ull * BS * DIM * 2;   // ~4.7MB
    const size_t needPart = (size_t)(WS_ROWM + 2 * BS * 8) * 4;         // ~0.5MB
    const bool  bf16path  = ws_size >= needBf16;
    int nsplit = (bf16path || ws_size >= needPart) ? 8 : 1;

    float*  rowM = ws + WS_ROWM;
    float*  rowS = ws + WS_ROWM + BS * nsplit;
    __bf16* Fhi  = (__bf16*)(ws + WS_FHI);
    __bf16* Flo  = (__bf16*)(ws + WS_FLO);

    scl_zero<<<6, 256, 0, stream>>>(ws);
    scl_classsum<<<BS / 64, 128, 0, stream>>>(F, T, clsSum);
    scl_counts<<<1, 256, 0, stream>>>(T, counts);
    if (bf16path) {
        scl_split<<<(BS * DIM) / 256, 256, 0, stream>>>(F, Fhi, Flo);
        scl_gemm_softmax_bf16<<<(NTILES * nsplit) / 4, 128, 0, stream>>>(Fhi, Flo, rowM, rowS, nsplit);
    } else {
        scl_gemm_softmax_f32<<<(NTILES * nsplit) / 4, 128, 0, stream>>>(F, rowM, rowS, nsplit);
    }
    scl_rowloss<<<BS / 8, 256, 0, stream>>>(F, T, clsSum, counts, rowM, rowS, lossSum, nsplit);
    scl_final<<<1, 32, 0, stream>>>(counts, lossSum, out);
}
